// QuantumRecurrentUnit_85418309583352
// MI455X (gfx1250) — matrix operand order repaired and checked by reading, not compiled, not run on hardware
//
#include <hip/hip_runtime.h>
#include <hip/hip_bf16.h>

typedef _Float16 v16h __attribute__((ext_vector_type(16)));
typedef _Float16 v8h  __attribute__((ext_vector_type(8)));
typedef float    v8f  __attribute__((ext_vector_type(8)));

#define NQ   6
#define DIM  64
#define TQ   1024
#define NF   16
#define BT   16          // batch columns per block
#define VSTR 72          // padded LDS row stride (halves) to break bank conflicts

// ---------------------------------------------------------------------------
// Setup: fold CNOT-chain permutation P and weight layer U = kron_q RY(w_q)
// into one 64x64 f16 matrix M with M[i][P[j]] = U[i][j].
// ---------------------------------------------------------------------------
__global__ void qru_setup(const float* __restrict__ w, _Float16* __restrict__ M) {
    int i = threadIdx.x;                       // output row, 0..63
    float c[NQ], s[NQ];
#pragma unroll
    for (int q = 0; q < NQ; ++q) {
        float a = 0.5f * w[q];
        s[q] = __sinf(a);
        c[q] = __cosf(a);
    }
    for (int j = 0; j < DIM; ++j) {
        // P = p0 ∘ p1 ∘ p2 ∘ p3 ∘ p4 (gather composition), p_q = CNOT(q, q+1)
        int p = j;
#pragma unroll
        for (int q = 4; q >= 0; --q) {
            if ((p >> (5 - q)) & 1) p ^= 1 << (4 - q);
        }
        float u = 1.0f;
#pragma unroll
        for (int q = 0; q < NQ; ++q) {
            int bi = (i >> (5 - q)) & 1;
            int bj = (j >> (5 - q)) & 1;
            u *= (bi == bj) ? c[q] : (bi ? s[q] : -s[q]);
        }
        M[i * DIM + p] = (_Float16)u;
    }
}

// ---------------------------------------------------------------------------
// Persistent recurrent kernel: block = 4 waves = 16 batch columns, loops t.
// ---------------------------------------------------------------------------
__global__ void __launch_bounds__(128) qru_main(const float* __restrict__ x,
                                                const _Float16* __restrict__ M,
                                                float* __restrict__ out) {
    __shared__ _Float16 Vt[BT * VSTR];     // product states, transposed: [col][K]
    __shared__ float cS[NQ * BT];          // cos(theta/2) [q][b]
    __shared__ float sS[NQ * BT];          // sin(theta/2) [q][b]
    __shared__ float partS[8 * NQ * BT];   // per-half-wave partials [g][q][b]

    const int tid  = threadIdx.x;
    const int lane = tid & 31;
    const int wv   = tid >> 5;             // wave id = output row-tile
    const int bg   = blockIdx.x * BT;      // global batch base

    // A fragments: rows 16*wv..16*wv+15 of M, K-chunks [0,32) and [32,64).
    // f16 A layout: lane L holds row L%16; halves j cover K = kbase + j (j<8)
    // and K = 16 + kbase + j (j>=8), kbase = (L>=16)*8. Loaded ONCE.
    const int arow = wv * 16 + (lane & 15);
    const int kbas = (lane >> 4) * 8;
    union { v16h v; v8h h[2]; } a0u, a1u, b0u, b1u;
    a0u.h[0] = *(const v8h*)(M + arow * DIM +  0 + kbas);
    a0u.h[1] = *(const v8h*)(M + arow * DIM + 16 + kbas);
    a1u.h[0] = *(const v8h*)(M + arow * DIM + 32 + kbas);
    a1u.h[1] = *(const v8h*)(M + arow * DIM + 48 + kbas);

    const int bq_b = tid / NQ;             // for tid < 96: batch column
    const int bq_q = tid % NQ;             // wire index
    float prev = 0.0f;                     // recurrent state lives in a register

    // Vt-build mapping: thread owns column vb and 8 consecutive rows blk*8..+7
    const int vb  = tid >> 3;              // 0..15
    const int blk = tid & 7;               // high 3 bits of state index

    // WMMA B/C layout constants
    const int col = lane & 15;
    const int hs  = (lane >> 4) * 16;      // K sub-block select for B
    const int grp = wv * 2 + (lane >> 4);  // 0..7: which 8-row slice of Y

    for (int t = 0; t < TQ; ++t) {
        // ---- phase 1: theta = prev + x[:, t, :6]; half-angle sin/cos ----
        if (tid < NQ * BT) {
            float th = prev + x[(size_t)(bg + bq_b) * TQ * NF + (size_t)t * NF + bq_q];
            float sn, cn;
            __sincosf(0.5f * th, &sn, &cn);
            cS[bq_q * BT + bq_b] = cn;
            sS[bq_q * BT + bq_b] = sn;
        }
        if (tid < BT && t + 1 < TQ) {
            __builtin_prefetch(x + (size_t)(bg + tid) * TQ * NF + (size_t)(t + 1) * NF, 0, 1);
        }
        __syncthreads();

        // ---- phase 2: product state V^T[vb][i], 8 rows per thread, one b128 store
        {
            float h = 1.0f;
#pragma unroll
            for (int q = 0; q < 3; ++q) {          // wires 0..2 <-> bits 5..3 = blk
                int bit = (blk >> (2 - q)) & 1;
                h *= bit ? sS[q * BT + vb] : cS[q * BT + vb];
            }
            float c3 = cS[3 * BT + vb], s3 = sS[3 * BT + vb];
            float c4 = cS[4 * BT + vb], s4 = sS[4 * BT + vb];
            float c5 = cS[5 * BT + vb], s5 = sS[5 * BT + vb];
            float h0 = h * c3, h1 = h * s3;
            float p00 = c4 * c5, p01 = c4 * s5, p10 = s4 * c5, p11 = s4 * s5;
            v8h hv;
            hv[0] = (_Float16)(h0 * p00); hv[1] = (_Float16)(h0 * p01);
            hv[2] = (_Float16)(h0 * p10); hv[3] = (_Float16)(h0 * p11);
            hv[4] = (_Float16)(h1 * p00); hv[5] = (_Float16)(h1 * p01);
            hv[6] = (_Float16)(h1 * p10); hv[7] = (_Float16)(h1 * p11);
            *(v8h*)(Vt + vb * VSTR + blk * 8) = hv;
        }
        __syncthreads();

        // ---- phase 3: Y(64x16) = M * V via 2 WMMAs per wave ----
        b0u.h[0] = *(const v8h*)(Vt + col * VSTR +  0 + hs / 2);
        b0u.h[1] = *(const v8h*)(Vt + col * VSTR + 16 + hs / 2);
        b1u.h[0] = *(const v8h*)(Vt + col * VSTR + 32 + hs / 2);
        b1u.h[1] = *(const v8h*)(Vt + col * VSTR + 48 + hs / 2);

        v8f acc = {0.f, 0.f, 0.f, 0.f, 0.f, 0.f, 0.f, 0.f};
        acc = __builtin_amdgcn_wmma_f32_16x16x32_f16(false, a0u.v, false, b0u.v,
                                                     (short)0, acc, false, false);
        acc = __builtin_amdgcn_wmma_f32_16x16x32_f16(false, a1u.v, false, b1u.v,
                                                     (short)0, acc, false, false);

        // ---- signed square-sums via Walsh tree over the 8 rows ----
        float y2[8];
#pragma unroll
        for (int r = 0; r < 8; ++r) y2[r] = acc[r] * acc[r];
        float t01 = y2[0] + y2[1], t23 = y2[2] + y2[3];
        float t45 = y2[4] + y2[5], t67 = y2[6] + y2[7];
        float u01 = y2[0] - y2[1], u23 = y2[2] - y2[3];
        float u45 = y2[4] - y2[5], u67 = y2[6] - y2[7];
        float a   = t01 + t23, bsum = t45 + t67;
        float S   = a + bsum;                        // plain sum
        float sb2 = a - bsum;                        // sign on bit2  (q=3)
        float sb1 = (t01 - t23) + (t45 - t67);       // sign on bit1  (q=4)
        float sb0 = (u01 + u23) + (u45 + u67);       // sign on bit0  (q=5)
        const int ibase = grp << 3;                  // fixed bits 5..3 of the rows
        float p0 = ((ibase >> 5) & 1) ? -S : S;      // q=0
        float p1 = ((ibase >> 4) & 1) ? -S : S;      // q=1
        float p2 = ((ibase >> 3) & 1) ? -S : S;      // q=2
        partS[(grp * NQ + 0) * BT + col] = p0;
        partS[(grp * NQ + 1) * BT + col] = p1;
        partS[(grp * NQ + 2) * BT + col] = p2;
        partS[(grp * NQ + 3) * BT + col] = sb2;
        partS[(grp * NQ + 4) * BT + col] = sb1;
        partS[(grp * NQ + 5) * BT + col] = sb0;
        __syncthreads();

        // ---- phase 4: reduce 8 slices, emit output, update recurrence ----
        if (tid < NQ * BT) {
            float o = 0.0f;
#pragma unroll
            for (int g = 0; g < 8; ++g) o += partS[(g * NQ + bq_q) * BT + bq_b];
            out[(size_t)(bg + bq_b) * TQ * NQ + (size_t)t * NQ + bq_q] = o;
            prev = o;
        }
        // no barrier needed here: next write to partS is after barrier #2,
        // next write to cS/sS is ordered by barrier #1 of the next iteration
    }
}

extern "C" void kernel_launch(void* const* d_in, const int* in_sizes, int n_in,
                              void* d_out, int out_size, void* d_ws, size_t ws_size,
                              hipStream_t stream) {
    const float* x = (const float*)d_in[0];     // [B, 1024, 16] f32
    const float* w = (const float*)d_in[1];     // [6] f32
    float* out     = (float*)d_out;             // [B, 1024, 6] f32
    _Float16* M    = (_Float16*)d_ws;           // 64*64 f16 folded gate matrix

    const int B = in_sizes[0] / (TQ * NF);      // 2048

    qru_setup<<<1, 64, 0, stream>>>(w, M);
    qru_main<<<B / BT, 128, 0, stream>>>(x, M, out);
}
